// GCN_21157008900499
// MI455X (gfx1250) — matrix operand order repaired and checked by reading, not compiled, not run on hardware
//
#include <hip/hip_runtime.h>
#include <hip/hip_bf16.h>

#define NNODES 100000
#define NEDGES 1600000
#define IN_FT  512
#define HID    128
#define OUTF   64

// Flip to 0 if the assembler rejects v_cvt_pk_bf16_f32 on gfx1250.
#define USE_ASM_CVT 1

typedef __attribute__((ext_vector_type(16))) __bf16 v16bf;
typedef __attribute__((ext_vector_type(8)))  float  v8f;
typedef __attribute__((ext_vector_type(4)))  float  v4f;  // native vector for NT loads

union BF16x16 {
  unsigned short s[16];
  unsigned       w[8];
  uint4          u[2];
  v16bf          v;
};

__device__ __forceinline__ unsigned short f2bf(float f) {
  union { float f; unsigned u; } a;
  a.f = f;
  unsigned u = a.u + 0x7FFFu + ((a.u >> 16) & 1u);  // round-to-nearest-even
  return (unsigned short)(u >> 16);
}

// Pack two f32 -> two bf16 in one dword. Native v_cvt_pk_bf16_f32 (one VALU op)
// via inline asm; clang-22 does not declare the builtin for gfx1250.
__device__ __forceinline__ unsigned bfpack2(float lo, float hi) {
#if USE_ASM_CVT
  unsigned r;
  asm("v_cvt_pk_bf16_f32 %0, %1, %2" : "=v"(r) : "v"(lo), "v"(hi));
  return r;
#else
  return (unsigned)f2bf(lo) | ((unsigned)f2bf(hi) << 16);
#endif
}

// ---------------- degree / normalization ----------------
__global__ void deg_init(float* __restrict__ deg) {
  int i = blockIdx.x * 256 + threadIdx.x;
  if (i < NNODES) deg[i] = 1.0f;  // self-loop weight
}

__global__ void deg_edges(const int* __restrict__ col, const float* __restrict__ w,
                          float* __restrict__ deg) {
  int e = blockIdx.x * 256 + threadIdx.x;  // grid sized exactly E
  atomicAdd(&deg[col[e]], w[e]);
}

__global__ void deg_to_dinv(float* __restrict__ deg) {
  int i = blockIdx.x * 256 + threadIdx.x;
  if (i < NNODES) {
    float d = deg[i];
    deg[i] = d > 0.0f ? rsqrtf(d) : 0.0f;
  }
}

// ---------------- pack weights into WMMA B-operand layout ----------------
// B (32x16 bf16) per-lane layout: lane = N%16 + 16*h, element v holds
// K = 8h + v (v<8) or 16 + 8h + (v-8) (v>=8). Packed buffer: [nt][kt][lane][16].
__global__ void pack_w1(const float* __restrict__ W1, unsigned short* __restrict__ w1p) {
  int tid  = blockIdx.x * 256 + threadIdx.x;  // 65536 total
  int v    = tid & 15;
  int lane = (tid >> 4) & 31;
  int kt   = (tid >> 9) & 15;
  int nt   = tid >> 13;
  int K = kt * 32 + ((lane >> 4) << 3) + (v & 7) + ((v >> 3) << 4);
  int N = nt * 16 + (lane & 15);
  w1p[tid] = f2bf(W1[K * HID + N]);
}

__global__ void pack_w2(const float* __restrict__ W2, unsigned short* __restrict__ w2p) {
  int tid  = blockIdx.x * 256 + threadIdx.x;  // 8192 total
  int v    = tid & 15;
  int lane = (tid >> 4) & 31;
  int kt   = (tid >> 9) & 3;
  int nt   = tid >> 11;
  int K = kt * 32 + ((lane >> 4) << 3) + (v & 7) + ((v >> 3) << 4);
  int N = nt * 16 + (lane & 15);
  w2p[tid] = f2bf(W2[K * OUTF + N]);
}

// ---------------- GEMM1: h1 = X @ W1  (bf16 WMMA, f32 accumulate) -----------
// Block = 8 waves, all on ntile = blockIdx.y; wave w -> mtile blockIdx.x*8+w.
// A (16x32 bf16): lane holds row M=lane%16; K runs [off8, off8+8) and
// [off8+16, off8+24) with off8 = 8*(lane>=16); converted from f32 in-register
// with v_cvt_pk_bf16_f32. X is streamed once -> non-temporal loads so the
// 192 MB L2 keeps h1/agg resident for the edge phase.
__global__ void gemm_x_w1(const float* __restrict__ x,
                          const unsigned short* __restrict__ w1p,
                          float* __restrict__ h1) {
  __shared__ uint4 Bs[16 * 32 * 2];  // 16 k-tiles of packed B, 16 KB
  const int nt   = blockIdx.y;
  const int lane = threadIdx.x & 31;
  const int wave = threadIdx.x >> 5;
  const uint4* wsrc = (const uint4*)(w1p + (size_t)nt * (16 * 32 * 16));
  for (int i = threadIdx.x; i < 16 * 32 * 2; i += 256) Bs[i] = wsrc[i];
  __syncthreads();
  const int mt = blockIdx.x * 8 + wave;
  if (mt < NNODES / 16) {
    const int   row  = mt * 16 + (lane & 15);
    const int   off8 = (lane >> 4) << 3;
    const float* xrow = x + (size_t)row * IN_FT;
    v8f acc = {};
#pragma unroll 4
    for (int kt = 0; kt < IN_FT / 32; ++kt) {
      const v4f* pa = (const v4f*)(xrow + kt * 32 + off8);
      v4f f0 = __builtin_nontemporal_load(pa + 0);
      v4f f1 = __builtin_nontemporal_load(pa + 1);
      const v4f* pb = (const v4f*)(xrow + kt * 32 + 16 + off8);
      v4f f2 = __builtin_nontemporal_load(pb + 0);
      v4f f3 = __builtin_nontemporal_load(pb + 1);
      BF16x16 A;
      A.w[0] = bfpack2(f0.x, f0.y);
      A.w[1] = bfpack2(f0.z, f0.w);
      A.w[2] = bfpack2(f1.x, f1.y);
      A.w[3] = bfpack2(f1.z, f1.w);
      A.w[4] = bfpack2(f2.x, f2.y);
      A.w[5] = bfpack2(f2.z, f2.w);
      A.w[6] = bfpack2(f3.x, f3.y);
      A.w[7] = bfpack2(f3.z, f3.w);
      BF16x16 B;
      B.u[0] = Bs[(kt * 32 + lane) * 2 + 0];
      B.u[1] = Bs[(kt * 32 + lane) * 2 + 1];
      acc = __builtin_amdgcn_wmma_f32_16x16x32_bf16(false, A.v, false, B.v,
                                                    (short)0, acc, false, false);
    }
    // C/D layout: VGPR v -> M = v + 8*(lane>=16), N = lane%16
    const int col = nt * 16 + (lane & 15);
    const int m0  = mt * 16 + ((lane >> 4) << 3);
    float* hp = h1 + (size_t)m0 * HID + col;
#pragma unroll
    for (int v = 0; v < 8; ++v) hp[(size_t)v * HID] = acc[v];
  }
}

// ---------------- GEMM2: h2 = relu(agg1+b1) @ W2 (A already bf16-packed) ----
__global__ void gemm_h_w2(const unsigned short* __restrict__ h1bf,
                          const unsigned short* __restrict__ w2p,
                          float* __restrict__ h2) {
  __shared__ uint4 Bs[4 * 32 * 2];  // 4 KB
  const int nt   = blockIdx.y;
  const int lane = threadIdx.x & 31;
  const int wave = threadIdx.x >> 5;
  const uint4* wsrc = (const uint4*)(w2p + (size_t)nt * (4 * 32 * 16));
  for (int i = threadIdx.x; i < 4 * 32 * 2; i += 256) Bs[i] = wsrc[i];
  __syncthreads();
  const int mt = blockIdx.x * 8 + wave;
  if (mt < NNODES / 16) {
    const int row  = mt * 16 + (lane & 15);
    const int off8 = (lane >> 4) << 3;
    const unsigned short* arow = h1bf + (size_t)row * HID;
    v8f acc = {};
#pragma unroll
    for (int kt = 0; kt < HID / 32; ++kt) {
      BF16x16 A;
      A.u[0] = *(const uint4*)(arow + kt * 32 + off8);
      A.u[1] = *(const uint4*)(arow + kt * 32 + 16 + off8);
      BF16x16 B;
      B.u[0] = Bs[(kt * 32 + lane) * 2 + 0];
      B.u[1] = Bs[(kt * 32 + lane) * 2 + 1];
      acc = __builtin_amdgcn_wmma_f32_16x16x32_bf16(false, A.v, false, B.v,
                                                    (short)0, acc, false, false);
    }
    const int col = nt * 16 + (lane & 15);
    const int m0  = mt * 16 + ((lane >> 4) << 3);
    float* hp = h2 + (size_t)m0 * OUTF + col;
#pragma unroll
    for (int v = 0; v < 8; ++v) hp[(size_t)v * OUTF] = acc[v];
  }
}

// ---------------- message passing (L2-resident gather + atomic scatter) -----
__global__ void self_init1(const float* __restrict__ h1, const float* __restrict__ dinv,
                           float* __restrict__ agg) {
  size_t tid = (size_t)blockIdx.x * 256 + threadIdx.x;  // N*HID total
  int i = (int)(tid >> 7);
  float di = dinv[i];
  agg[tid] = di * di * h1[tid];
}

__global__ void scatter1(const int* __restrict__ row, const int* __restrict__ col,
                         const float* __restrict__ w, const float* __restrict__ dinv,
                         const float* __restrict__ h1, float* __restrict__ agg) {
  int e    = blockIdx.x * 8 + (threadIdx.x >> 5);  // one wave per edge
  int lane = threadIdx.x & 31;
  int r = row[e], c = col[e];
  float nrm = dinv[r] * w[e] * dinv[c];
  float4 v = *((const float4*)(h1 + (size_t)r * HID) + lane);  // 4 feats/lane
  float* ap = agg + (size_t)c * HID + lane * 4;
  atomicAdd(ap + 0, nrm * v.x);
  atomicAdd(ap + 1, nrm * v.y);
  atomicAdd(ap + 2, nrm * v.z);
  atomicAdd(ap + 3, nrm * v.w);
}

// 2 features per thread: float2 in -> one packed bf16 dword out.
__global__ void bias_relu_pack(const float* __restrict__ agg, const float* __restrict__ b1,
                               unsigned* __restrict__ h1bf) {
  size_t tid = (size_t)blockIdx.x * 256 + threadIdx.x;  // N*HID/2 total
  int fp = (int)(tid & (HID / 2 - 1));                  // feature pair
  float2 v = ((const float2*)agg)[tid];
  float a = v.x + b1[fp * 2 + 0];
  float b = v.y + b1[fp * 2 + 1];
  h1bf[tid] = bfpack2(a > 0.0f ? a : 0.0f, b > 0.0f ? b : 0.0f);
}

__global__ void self_init2(const float* __restrict__ h2, const float* __restrict__ dinv,
                           float* __restrict__ out) {
  size_t tid = (size_t)blockIdx.x * 256 + threadIdx.x;  // N*OUTF total
  int i = (int)(tid >> 6);
  float di = dinv[i];
  out[tid] = di * di * h2[tid];
}

__global__ void scatter2(const int* __restrict__ row, const int* __restrict__ col,
                         const float* __restrict__ w, const float* __restrict__ dinv,
                         const float* __restrict__ h2, float* __restrict__ out) {
  int e    = blockIdx.x * 8 + (threadIdx.x >> 5);
  int lane = threadIdx.x & 31;
  int r = row[e], c = col[e];
  float nrm = dinv[r] * w[e] * dinv[c];
  float2 v = *((const float2*)(h2 + (size_t)r * OUTF) + lane);  // 2 feats/lane
  float* ap = out + (size_t)c * OUTF + lane * 2;
  atomicAdd(ap + 0, nrm * v.x);
  atomicAdd(ap + 1, nrm * v.y);
}

// ---------------- bias + log_softmax, one wave per node (wave32) ------------
__global__ void bias_logsoftmax(float* __restrict__ out, const float* __restrict__ b2) {
  int i    = blockIdx.x * 8 + (threadIdx.x >> 5);
  int lane = threadIdx.x & 31;
  float2* p = (float2*)(out + (size_t)i * OUTF) + lane;
  float2 v = *p;
  v.x += b2[lane * 2 + 0];
  v.y += b2[lane * 2 + 1];
  float m = fmaxf(v.x, v.y);
#pragma unroll
  for (int d = 16; d > 0; d >>= 1) m = fmaxf(m, __shfl_xor(m, d, 32));
  float s = __expf(v.x - m) + __expf(v.y - m);
#pragma unroll
  for (int d = 16; d > 0; d >>= 1) s += __shfl_xor(s, d, 32);
  float lse = m + __logf(s);
  v.x -= lse;
  v.y -= lse;
  *p = v;
}

extern "C" void kernel_launch(void* const* d_in, const int* in_sizes, int n_in,
                              void* d_out, int out_size, void* d_ws, size_t ws_size,
                              hipStream_t stream) {
  const float* x  = (const float*)d_in[0];
  const int*   ei = (const int*)d_in[1];
  const float* ew = (const float*)d_in[2];
  const float* W1 = (const float*)d_in[3];
  const float* b1 = (const float*)d_in[4];
  const float* W2 = (const float*)d_in[5];
  const float* b2 = (const float*)d_in[6];
  float* out = (float*)d_out;
  const int* erow = ei;            // edge_index[0]
  const int* ecol = ei + NEDGES;   // edge_index[1]

  char* ws = (char*)d_ws;
  size_t off = 0;
  auto alloc = [&](size_t bytes) {
    size_t o = off;
    off += (bytes + 255) & ~(size_t)255;
    return o;
  };
  float*          dinv = (float*)(ws + alloc((size_t)NNODES * 4));
  unsigned short* w1p  = (unsigned short*)(ws + alloc((size_t)IN_FT * HID * 2));
  unsigned short* w2p  = (unsigned short*)(ws + alloc((size_t)HID * OUTF * 2));
  float*          h1   = (float*)(ws + alloc((size_t)NNODES * HID * 4));
  float*          agg1 = (float*)(ws + alloc((size_t)NNODES * HID * 4));
  unsigned short* h1bf = (unsigned short*)(ws + alloc((size_t)NNODES * HID * 2));
  float*          h2   = (float*)(ws + alloc((size_t)NNODES * OUTF * 4));
  (void)ws_size; (void)in_sizes; (void)n_in; (void)out_size;

  const int nblk = (NNODES + 255) / 256;

  deg_init<<<nblk, 256, 0, stream>>>(dinv);
  deg_edges<<<NEDGES / 256, 256, 0, stream>>>(ecol, ew, dinv);
  deg_to_dinv<<<nblk, 256, 0, stream>>>(dinv);

  pack_w1<<<(IN_FT * HID) / 256, 256, 0, stream>>>(W1, w1p);
  pack_w2<<<(HID * OUTF) / 256, 256, 0, stream>>>(W2, w2p);

  // GEMM1: 6250 m-tiles / 8 waves per block
  gemm_x_w1<<<dim3((NNODES / 16 + 7) / 8, HID / 16), 256, 0, stream>>>(x, w1p, h1);

  self_init1<<<(NNODES * HID) / 256, 256, 0, stream>>>(h1, dinv, agg1);
  scatter1<<<NEDGES / 8, 256, 0, stream>>>(erow, ecol, ew, dinv, h1, agg1);
  bias_relu_pack<<<(NNODES * HID / 2) / 256, 256, 0, stream>>>(agg1, b1,
                                                               (unsigned*)h1bf);

  gemm_h_w2<<<dim3((NNODES / 16 + 7) / 8, OUTF / 16), 256, 0, stream>>>(h1bf, w2p, h2);

  self_init2<<<(NNODES * OUTF) / 256, 256, 0, stream>>>(h2, dinv, out);
  scatter2<<<NEDGES / 8, 256, 0, stream>>>(erow, ecol, ew, dinv, h2, out);

  bias_logsoftmax<<<NNODES / 8, 256, 0, stream>>>(out, b2);
}
